// RNN_72430328479763
// MI455X (gfx1250) — matrix operand order repaired and checked by reading, not compiled, not run on hardware
//
#include <hip/hip_runtime.h>

// ---------------------------------------------------------------------------
// Tacotron-style attention decoder for MI455X (gfx1250, wave32, WMMA bf16)
// ---------------------------------------------------------------------------
#define Bb   32
#define Ss   512
#define Tt   512
#define ENCd 512
#define ATTd 128
#define PHd  256
#define POd  256
#define DECd 1024
#define Md   128

typedef __attribute__((ext_vector_type(16))) __bf16 bf16x16;
typedef __attribute__((ext_vector_type(8)))  float  f32x8;
typedef unsigned short u16;

struct Frag { union { uint4 u[2]; bf16x16 v; }; };

__device__ inline f32x8 wmma_bf16(const Frag& a, const Frag& b, f32x8 c) {
  return __builtin_amdgcn_wmma_f32_16x16x32_bf16(
      false, a.v, false, b.v, (short)0, c, false, false);
}

// A matrix 16x32 bf16, row-major [row][k] with leading dim ldk (mult of 8).
__device__ inline void loadA16(Frag& f, const u16* base, int row0, int k0,
                               int ldk, int lane) {
  const u16* p = base + (size_t)(row0 + (lane & 15)) * ldk +
                 (k0 + ((lane >> 4) << 4));
  f.u[0] = ((const uint4*)p)[0];
  f.u[1] = ((const uint4*)p)[1];
}
// B matrix 32x16 bf16 from weights stored [out][in] (ldk = in-dim, mult of 8).
__device__ inline void loadB16(Frag& f, const u16* base, int n0, int k0,
                               int ldk, int lane) {
  const u16* p = base + (size_t)(n0 + (lane & 15)) * ldk +
                 (k0 + ((lane >> 4) << 4));
  f.u[0] = ((const uint4*)p)[0];
  f.u[1] = ((const uint4*)p)[1];
}

__device__ inline float bf2f(u16 b) {
  return __builtin_bit_cast(float, (unsigned)b << 16);
}
__device__ inline u16 f2bf(float f) {  // round-to-nearest-even
  unsigned u = __builtin_bit_cast(unsigned, f);
  unsigned r = u + 0x7FFFu + ((u >> 16) & 1u);
  return (u16)(r >> 16);
}

#if __has_builtin(__builtin_amdgcn_tanhf)
__device__ inline float tanh_fast(float x) { return __builtin_amdgcn_tanhf(x); }
#else
__device__ inline float tanh_fast(float x) {
  float r;
  asm volatile("v_tanh_f32 %0, %1" : "=v"(r) : "v"(x));
  return r;
}
#endif
__device__ inline float sigmoid_fast(float x) {
  return __builtin_amdgcn_rcpf(1.f + __builtin_amdgcn_exp2f(-x * 1.44269504089f));
}

// ---------------------------------------------------------------------------
// Prep kernels: f32 -> bf16 conversions / transposes into workspace
// ---------------------------------------------------------------------------
__global__ void cvt_bf16_kernel(const float* __restrict__ s, u16* __restrict__ d,
                                size_t n) {
  for (size_t i = (size_t)blockIdx.x * blockDim.x + threadIdx.x; i < n;
       i += (size_t)gridDim.x * blockDim.x)
    d[i] = f2bf(s[i]);
}
// s[R][C] f32  ->  d[C][R] bf16   ("[in][out]" weights -> WMMA-B "[out][in]")
__global__ void transpose_bf16_kernel(const float* __restrict__ s,
                                      u16* __restrict__ d, int R, int C) {
  size_t n = (size_t)R * C;
  for (size_t i = (size_t)blockIdx.x * blockDim.x + threadIdx.x; i < n;
       i += (size_t)gridDim.x * blockDim.x) {
    int r = (int)(i / C), c = (int)(i % C);
    d[(size_t)c * R + r] = f2bf(s[i]);
  }
}
// teacher forcing shift: row (t*32+b) <- t==0 ? 0 : target_mels[b][t-1][:]
__global__ void build_prevmel_kernel(const float* __restrict__ tm,
                                     u16* __restrict__ d) {
  size_t n = (size_t)Tt * Bb * Md;
  for (size_t i = (size_t)blockIdx.x * blockDim.x + threadIdx.x; i < n;
       i += (size_t)gridDim.x * blockDim.x) {
    int m = (int)(i & (Md - 1));
    size_t rb = i >> 7;
    int b = (int)(rb & 31), t = (int)(rb >> 5);
    float v = (t == 0) ? 0.f : tm[((size_t)b * Tt + (t - 1)) * Md + m];
    d[i] = f2bf(v);
  }
}

// ---------------------------------------------------------------------------
// PreNet for ALL timesteps (parallel, WMMA):  X[T*B][256] bf16
// block = 128 threads (4 waves) handles 16 rows. One shared A fragment feeds
// 4 independent WMMA accumulator chains per K-step (hides WMMA RAW latency).
// ---------------------------------------------------------------------------
__global__ void prenet_kernel(const u16* __restrict__ prevMelB,
                              const u16* __restrict__ W1T,  // [256][128]
                              const u16* __restrict__ W2T,  // [256][256]
                              const float* __restrict__ b1,
                              const float* __restrict__ b2,
                              u16* __restrict__ X) {
  extern __shared__ char sm[];
  u16* sW   = (u16*)sm;              // up to 256*256 bf16 = 128KB
  u16* sHid = (u16*)(sm + 131072);   // 16*256 bf16
  u16* sA   = (u16*)(sm + 139264);   // 16*128 bf16
  int tid = threadIdx.x, lane = tid & 31, wave = tid >> 5;
  int row0 = blockIdx.x * 16;
  int mh = (lane >> 4) * 8;

  for (int i = tid; i < 16 * 128; i += 128)
    sA[i] = prevMelB[(size_t)row0 * 128 + i];
  for (int i = tid; i < 256 * 128; i += 128) sW[i] = W1T[i];
  __syncthreads();

  {  // layer 1: 16 N-tiles / 4 waves, 4 independent chains per wave
    f32x8 acc[4] = {};
    for (int kk = 0; kk < 4; ++kk) {
      Frag A;
      loadA16(A, sA, 0, kk * 32, 128, lane);
#pragma unroll
      for (int q = 0; q < 4; ++q) {
        Frag Bw;
        loadB16(Bw, sW, (wave * 4 + q) * 16, kk * 32, 128, lane);
        acc[q] = wmma_bf16(A, Bw, acc[q]);
      }
    }
#pragma unroll
    for (int q = 0; q < 4; ++q) {
      int col = (wave * 4 + q) * 16 + (lane & 15);
      float bb = b1[col];
#pragma unroll
      for (int g = 0; g < 8; ++g)
        sHid[(mh + g) * 256 + col] = f2bf(fmaxf(acc[q][g] + bb, 0.f));
    }
  }
  __syncthreads();
  for (int i = tid; i < 256 * 256; i += 128) sW[i] = W2T[i];
  __syncthreads();

  {  // layer 2
    f32x8 acc[4] = {};
    for (int kk = 0; kk < 8; ++kk) {
      Frag A;
      loadA16(A, sHid, 0, kk * 32, 256, lane);
#pragma unroll
      for (int q = 0; q < 4; ++q) {
        Frag Bw;
        loadB16(Bw, sW, (wave * 4 + q) * 16, kk * 32, 256, lane);
        acc[q] = wmma_bf16(A, Bw, acc[q]);
      }
    }
#pragma unroll
    for (int q = 0; q < 4; ++q) {
      int col = (wave * 4 + q) * 16 + (lane & 15);
      float bb = b2[col];
#pragma unroll
      for (int g = 0; g < 8; ++g)
        X[(size_t)(row0 + mh + g) * 256 + col] =
            f2bf(fmaxf(acc[q][g] + bb, 0.f));
    }
  }
}

// ---------------------------------------------------------------------------
// enc_proj[B*S][128] = encB[B*S][512] @ WeT^T + att_be   (parallel, WMMA)
// block = 256 threads (8 waves) handles 32 rows; each wave does 2 N-adjacent
// tiles sharing the A fragment -> 2 independent WMMA chains per K-step.
// ---------------------------------------------------------------------------
__global__ void encproj_kernel(const u16* __restrict__ encB,
                               const u16* __restrict__ WeT,   // [128][512]
                               const float* __restrict__ att_be,
                               float* __restrict__ encProj) {
  extern __shared__ char sm[];
  u16* sW = (u16*)sm;                 // 128*512 bf16 = 128KB
  u16* sA = (u16*)(sm + 131072);      // 32*512 bf16 = 32KB
  int tid = threadIdx.x, lane = tid & 31, wave = tid >> 5;
  size_t row0 = (size_t)blockIdx.x * 32;

  for (int i = tid; i < 32 * 512; i += 256) sA[i] = encB[row0 * 512 + i];
  for (int i = tid; i < 128 * 512; i += 256) sW[i] = WeT[i];
  __syncthreads();

  int mt = wave >> 2, ntp = (wave & 3) * 2;
  f32x8 acc0 = {}, acc1 = {};
  for (int kk = 0; kk < 16; ++kk) {
    Frag A, B0, B1;
    loadA16(A, sA, mt * 16, kk * 32, 512, lane);
    loadB16(B0, sW, ntp * 16, kk * 32, 512, lane);
    loadB16(B1, sW, (ntp + 1) * 16, kk * 32, 512, lane);
    acc0 = wmma_bf16(A, B0, acc0);
    acc1 = wmma_bf16(A, B1, acc1);
  }
  int mhh = (lane >> 4) * 8;
  int col0 = ntp * 16 + (lane & 15), col1 = (ntp + 1) * 16 + (lane & 15);
  float bb0 = att_be[col0], bb1 = att_be[col1];
#pragma unroll
  for (int g = 0; g < 8; ++g) {
    encProj[(row0 + mt * 16 + mhh + g) * 128 + col0] = acc0[g] + bb0;
    encProj[(row0 + mt * 16 + mhh + g) * 128 + col1] = acc1[g] + bb1;
  }
}

// ---------------------------------------------------------------------------
// Sequential decoder: one persistent workgroup, 1024 threads = 32 wave32.
// LDS A-matrix hA[32][1792] bf16 = [ x(0..255) | ctx(256..767) | h(768..1791) ]
// so GRU r/z gates accumulate over the whole concat in ONE WMMA K-loop.
// ---------------------------------------------------------------------------
__global__ void __launch_bounds__(1024, 1)
decoder_kernel(const u16* __restrict__ encB,      // [B*S][512] bf16
               const float* __restrict__ encProj, // [B*S][128] f32
               const u16* __restrict__ X,         // [T*B][256] bf16
               const u16* __restrict__ WihB,      // [3072][768] bf16
               const u16* __restrict__ WhhB,      // [3072][1024] bf16
               const u16* __restrict__ WdT,       // [128][1024] bf16
               const u16* __restrict__ OutWT,     // [128][1536] bf16
               const float* __restrict__ att_bd,
               const float* __restrict__ att_v,
               const float* __restrict__ att_vb,
               const float* __restrict__ gru_bih,
               const float* __restrict__ gru_bhh,
               const float* __restrict__ out_b,
               const float* __restrict__ stop_w,
               const float* __restrict__ stop_b,
               float* __restrict__ out) {
  extern __shared__ char sm[];
  u16*   hA      = (u16*)sm;                 // 32*1792 bf16 = 112KB
  float* dprojL  = (float*)(sm + 114688);    // 32*128 f32 (K lower half)
  float* dprojL2 = (float*)(sm + 131072);    // 32*128 f32 (K upper half)
  float* awL     = (float*)(sm + 147456);    // 32*512 f32 (scores -> softmax)
  float* vL      = (float*)(sm + 212992);    // 128 f32

  const int tid = threadIdx.x, lane = tid & 31, wave = tid >> 5;
  const int mh = (lane >> 4) * 8;
  const size_t melOff = 0;
  const size_t stopOff = (size_t)Bb * Tt * Md;
  const size_t attnOff = stopOff + (size_t)Bb * Tt;
  const float vb = att_vb[0];

  for (int i = tid; i < 32 * 1792; i += 1024) hA[i] = 0;  // x=ctx=h=0
  if (tid < 128) vL[tid] = att_v[tid];
  __syncthreads();

  for (int t = 0; t < Tt; ++t) {
    // ---- (a) stage prenet output x_t into hA[:, 0:256] ----
    {
      int e = tid * 8;  // 8192 bf16, 8 contiguous per thread
      int b = e >> 8, k = e & 255;
      *(uint4*)&hA[b * 1792 + k] =
          *(const uint4*)&X[((size_t)t * 32 + b) * 256 + k];
    }
    __syncthreads();

    // ---- (b) dproj = h @ Wd + bd ; split-K across ALL 32 waves, two
    //          independent WMMA chains per wave ----
    {
      int w2 = wave & 15;
      int mt = w2 >> 3, nt = w2 & 7;
      int kb = (wave < 16) ? 0 : 16;
      f32x8 acc0 = {}, acc1 = {};
      for (int kk = kb; kk < kb + 16; kk += 2) {
        Frag A0, B0, A1, B1;
        loadA16(A0, hA, mt * 16, 768 + kk * 32, 1792, lane);
        loadB16(B0, WdT, nt * 16, kk * 32, 1024, lane);
        loadA16(A1, hA, mt * 16, 768 + (kk + 1) * 32, 1792, lane);
        loadB16(B1, WdT, nt * 16, (kk + 1) * 32, 1024, lane);
        acc0 = wmma_bf16(A0, B0, acc0);
        acc1 = wmma_bf16(A1, B1, acc1);
      }
      f32x8 acc = acc0 + acc1;
      int col = nt * 16 + (lane & 15);
      float bb = (wave < 16) ? att_bd[col] : 0.f;
      float* dst = (wave < 16) ? dprojL : dprojL2;
#pragma unroll
      for (int g = 0; g < 8; ++g)
        dst[(mt * 16 + mh + g) * 128 + col] = acc[g] + bb;
    }
    __syncthreads();

    // ---- (c) scores[b][s] = tanh(enc_proj + dproj) . v + vb ----
    for (int idx = tid; idx < Bb * Ss; idx += 1024) {
      int b = idx >> 9, s = idx & 511;
      const float4* ep = (const float4*)(encProj + ((size_t)(b * Ss + s)) * 128);
      const float4* dp = (const float4*)(dprojL + b * 128);
      const float4* dq = (const float4*)(dprojL2 + b * 128);
      const float4* vv = (const float4*)vL;
      float sc = vb;
      for (int q = 0; q < 32; ++q) {
        float4 e4 = ep[q], d4 = dp[q], f4 = dq[q], v4 = vv[q];
        sc += tanh_fast(e4.x + d4.x + f4.x) * v4.x;
        sc += tanh_fast(e4.y + d4.y + f4.y) * v4.y;
        sc += tanh_fast(e4.z + d4.z + f4.z) * v4.z;
        sc += tanh_fast(e4.w + d4.w + f4.w) * v4.w;
      }
      awL[b * Ss + s] = sc;
    }
    __syncthreads();

    // ---- (d) softmax per row (one wave per batch row) + emit attn ----
    {
      int b = wave;
      float mx = -1e30f;
      for (int s = lane; s < Ss; s += 32) mx = fmaxf(mx, awL[b * Ss + s]);
#pragma unroll
      for (int o = 16; o; o >>= 1) mx = fmaxf(mx, __shfl_xor(mx, o, 32));
      float sum = 0.f;
      for (int s = lane; s < Ss; s += 32) {
        float e = __builtin_amdgcn_exp2f((awL[b * Ss + s] - mx) * 1.44269504f);
        awL[b * Ss + s] = e;
        sum += e;
      }
#pragma unroll
      for (int o = 16; o; o >>= 1) sum += __shfl_xor(sum, o, 32);
      float inv = __builtin_amdgcn_rcpf(sum);
      for (int s = lane; s < Ss; s += 32) {
        float a = awL[b * Ss + s] * inv;
        awL[b * Ss + s] = a;
        out[attnOff + ((size_t)b * Tt + t) * Ss + s] = a;
      }
    }
    __syncthreads();

    // ---- (e) context[b] = aw[b] @ enc[b]  ->  hA[:, 256:768] bf16 ----
    {
      int b = wave;
      int e0 = lane * 16;
      float c[16];
#pragma unroll
      for (int i = 0; i < 16; ++i) c[i] = 0.f;
      const u16* eb = encB + ((size_t)b * Ss) * ENCd + e0;
      for (int s = 0; s < Ss; ++s) {
        float a = awL[b * Ss + s];
        const uint4* p = (const uint4*)(eb + (size_t)s * ENCd);
        uint4 q0 = p[0], q1 = p[1];
        unsigned w;
        w = q0.x; c[0] += a * __builtin_bit_cast(float, w << 16);
                  c[1] += a * __builtin_bit_cast(float, w & 0xFFFF0000u);
        w = q0.y; c[2] += a * __builtin_bit_cast(float, w << 16);
                  c[3] += a * __builtin_bit_cast(float, w & 0xFFFF0000u);
        w = q0.z; c[4] += a * __builtin_bit_cast(float, w << 16);
                  c[5] += a * __builtin_bit_cast(float, w & 0xFFFF0000u);
        w = q0.w; c[6] += a * __builtin_bit_cast(float, w << 16);
                  c[7] += a * __builtin_bit_cast(float, w & 0xFFFF0000u);
        w = q1.x; c[8] += a * __builtin_bit_cast(float, w << 16);
                  c[9] += a * __builtin_bit_cast(float, w & 0xFFFF0000u);
        w = q1.y; c[10] += a * __builtin_bit_cast(float, w << 16);
                  c[11] += a * __builtin_bit_cast(float, w & 0xFFFF0000u);
        w = q1.z; c[12] += a * __builtin_bit_cast(float, w << 16);
                  c[13] += a * __builtin_bit_cast(float, w & 0xFFFF0000u);
        w = q1.w; c[14] += a * __builtin_bit_cast(float, w << 16);
                  c[15] += a * __builtin_bit_cast(float, w & 0xFFFF0000u);
      }
      union { u16 s[16]; uint4 q[2]; } ct;
#pragma unroll
      for (int i = 0; i < 16; ++i) ct.s[i] = f2bf(c[i]);
      uint4* dst = (uint4*)&hA[b * 1792 + 256 + e0];
      dst[0] = ct.q[0];
      dst[1] = ct.q[1];
    }
    __syncthreads();

    // ---- (f) GRU: each wave owns 2 column tiles of the 1024 hidden cols.
    //          6 independent WMMAs per K-step share two A fragments. ----
    float hkeep[2][2][8];
    for (int jj = 0; jj < 2; ++jj) {
      int col0 = (wave * 2 + jj) * 16;
      f32x8 aR[2] = {}, aZ[2] = {}, aN[2] = {}, aH[2] = {};
      for (int kk = 0; kk < 56; ++kk) {  // K = 256(x)+512(ctx)+1024(h) = 1792
        int k = kk * 32;
        Frag A0, A1, Br, Bz, Bn;
        loadA16(A0, hA, 0, k, 1792, lane);
        loadA16(A1, hA, 16, k, 1792, lane);
        if (k < 768) {
          loadB16(Br, WihB, col0, k, 768, lane);
          loadB16(Bz, WihB + (size_t)1024 * 768, col0, k, 768, lane);
          loadB16(Bn, WihB + (size_t)2048 * 768, col0, k, 768, lane);
          __builtin_prefetch(WihB + (size_t)(col0 + (lane & 15)) * 768 + k + 32,
                             0, 0);
        } else {
          int kh = k - 768;
          loadB16(Br, WhhB, col0, kh, 1024, lane);
          loadB16(Bz, WhhB + (size_t)1024 * 1024, col0, kh, 1024, lane);
          loadB16(Bn, WhhB + (size_t)2048 * 1024, col0, kh, 1024, lane);
          __builtin_prefetch(
              WhhB + (size_t)(col0 + (lane & 15)) * 1024 + kh + 32, 0, 0);
        }
        aR[0] = wmma_bf16(A0, Br, aR[0]);
        aR[1] = wmma_bf16(A1, Br, aR[1]);
        aZ[0] = wmma_bf16(A0, Bz, aZ[0]);
        aZ[1] = wmma_bf16(A1, Bz, aZ[1]);
        if (k < 768) {  // inn over [x|ctx]
          aN[0] = wmma_bf16(A0, Bn, aN[0]);
          aN[1] = wmma_bf16(A1, Bn, aN[1]);
        } else {        // hn over h
          aH[0] = wmma_bf16(A0, Bn, aH[0]);
          aH[1] = wmma_bf16(A1, Bn, aH[1]);
        }
      }
      int col = col0 + (lane & 15);
      float br = gru_bih[col] + gru_bhh[col];
      float bz = gru_bih[1024 + col] + gru_bhh[1024 + col];
      float bin_ = gru_bih[2048 + col];
      float bhn = gru_bhh[2048 + col];
#pragma unroll
      for (int mt = 0; mt < 2; ++mt)
#pragma unroll
        for (int g = 0; g < 8; ++g) {
          int m = mt * 16 + mh + g;
          float hold = bf2f(hA[m * 1792 + 768 + col]);
          float r = sigmoid_fast(aR[mt][g] + br);
          float z = sigmoid_fast(aZ[mt][g] + bz);
          float n = tanh_fast(aN[mt][g] + bin_ + r * (aH[mt][g] + bhn));
          hkeep[jj][mt][g] = (1.f - z) * n + z * hold;
        }
    }
    __syncthreads();  // everyone done reading old h
#pragma unroll
    for (int jj = 0; jj < 2; ++jj) {
      int col = (wave * 2 + jj) * 16 + (lane & 15);
#pragma unroll
      for (int mt = 0; mt < 2; ++mt)
#pragma unroll
        for (int g = 0; g < 8; ++g)
          hA[(mt * 16 + mh + g) * 1792 + 768 + col] = f2bf(hkeep[jj][mt][g]);
    }
    __syncthreads();

    // ---- (g) mel = [h_new | ctx] @ out_w + out_b ; stop dot ----
    if (wave < 16) {
      int mt = wave >> 3, nt = wave & 7;
      f32x8 acc0 = {}, acc1 = {};
      for (int kk = 0; kk < 48; kk += 2) {  // 1024(h) + 512(ctx), 2 chains
        Frag A0, B0, A1, B1;
        if (kk < 32) {
          loadA16(A0, hA, mt * 16, 768 + kk * 32, 1792, lane);
          loadB16(B0, OutWT, nt * 16, kk * 32, 1536, lane);
        } else {
          int k2 = (kk - 32) * 32;
          loadA16(A0, hA, mt * 16, 256 + k2, 1792, lane);
          loadB16(B0, OutWT, nt * 16, 1024 + k2, 1536, lane);
        }
        if (kk + 1 < 32) {
          loadA16(A1, hA, mt * 16, 768 + (kk + 1) * 32, 1792, lane);
          loadB16(B1, OutWT, nt * 16, (kk + 1) * 32, 1536, lane);
        } else {
          int k2 = (kk + 1 - 32) * 32;
          loadA16(A1, hA, mt * 16, 256 + k2, 1792, lane);
          loadB16(B1, OutWT, nt * 16, 1024 + k2, 1536, lane);
        }
        acc0 = wmma_bf16(A0, B0, acc0);
        acc1 = wmma_bf16(A1, B1, acc1);
      }
      f32x8 acc = acc0 + acc1;
      int col = nt * 16 + (lane & 15);
      float bb = out_b[col];
#pragma unroll
      for (int g = 0; g < 8; ++g) {
        int b = mt * 16 + mh + g;
        out[melOff + ((size_t)b * Tt + t) * Md + col] = acc[g] + bb;
      }
    } else if (wave == 16) {
      int b = lane;
      float s = stop_b[0];
      for (int k = 0; k < 1024; ++k)
        s += bf2f(hA[b * 1792 + 768 + k]) * stop_w[k];
      for (int e = 0; e < 512; ++e)
        s += bf2f(hA[b * 1792 + 256 + e]) * stop_w[1024 + e];
      out[stopOff + (size_t)b * Tt + t] = s;
    }
    __syncthreads();
  }
}

// ---------------------------------------------------------------------------
extern "C" void kernel_launch(void* const* d_in, const int* in_sizes, int n_in,
                              void* d_out, int out_size, void* d_ws,
                              size_t ws_size, hipStream_t stream) {
  const float* enc      = (const float*)d_in[0];
  const float* tmels    = (const float*)d_in[1];
  const float* pn_w1    = (const float*)d_in[2];
  const float* pn_b1    = (const float*)d_in[3];
  const float* pn_w2    = (const float*)d_in[4];
  const float* pn_b2    = (const float*)d_in[5];
  const float* att_we   = (const float*)d_in[6];
  const float* att_be   = (const float*)d_in[7];
  const float* att_wd   = (const float*)d_in[8];
  const float* att_bd   = (const float*)d_in[9];
  const float* att_v    = (const float*)d_in[10];
  const float* att_vb   = (const float*)d_in[11];
  const float* gru_wih  = (const float*)d_in[12];
  const float* gru_whh  = (const float*)d_in[13];
  const float* gru_bih  = (const float*)d_in[14];
  const float* gru_bhh  = (const float*)d_in[15];
  const float* out_w    = (const float*)d_in[16];
  const float* out_b    = (const float*)d_in[17];
  const float* stop_w   = (const float*)d_in[18];
  const float* stop_b   = (const float*)d_in[19];
  float* out = (float*)d_out;
  char* ws = (char*)d_ws;

  size_t o = 0;
  auto carve = [&](size_t bytes) -> void* {
    void* p = ws + o;
    o = (o + bytes + 255) & ~(size_t)255;
    return p;
  };
  u16*   encB     = (u16*)carve((size_t)Bb * Ss * ENCd * 2);
  float* encProj  = (float*)carve((size_t)Bb * Ss * ATTd * 4);
  u16*   X        = (u16*)carve((size_t)Tt * Bb * POd * 2);
  u16*   prevMelB = (u16*)carve((size_t)Tt * Bb * Md * 2);
  u16*   WihB     = (u16*)carve((size_t)3 * DECd * (POd + ENCd) * 2);
  u16*   WhhB     = (u16*)carve((size_t)3 * DECd * DECd * 2);
  u16*   WdT      = (u16*)carve((size_t)ATTd * DECd * 2);
  u16*   WeT      = (u16*)carve((size_t)ATTd * ENCd * 2);
  u16*   OutWT    = (u16*)carve((size_t)Md * (DECd + ENCd) * 2);
  u16*   PnW1T    = (u16*)carve((size_t)PHd * Md * 2);
  u16*   PnW2T    = (u16*)carve((size_t)POd * PHd * 2);

  hipFuncSetAttribute((const void*)prenet_kernel,
                      hipFuncAttributeMaxDynamicSharedMemorySize, 143360);
  hipFuncSetAttribute((const void*)encproj_kernel,
                      hipFuncAttributeMaxDynamicSharedMemorySize, 163840);
  hipFuncSetAttribute((const void*)decoder_kernel,
                      hipFuncAttributeMaxDynamicSharedMemorySize, 213504);

  dim3 cb(256), cg(4096);
  cvt_bf16_kernel<<<cg, cb, 0, stream>>>(enc, encB, (size_t)Bb * Ss * ENCd);
  build_prevmel_kernel<<<cg, cb, 0, stream>>>(tmels, prevMelB);
  cvt_bf16_kernel<<<cg, cb, 0, stream>>>(gru_wih, WihB,
                                         (size_t)3 * DECd * (POd + ENCd));
  cvt_bf16_kernel<<<cg, cb, 0, stream>>>(gru_whh, WhhB,
                                         (size_t)3 * DECd * DECd);
  transpose_bf16_kernel<<<cg, cb, 0, stream>>>(att_wd, WdT, DECd, ATTd);
  transpose_bf16_kernel<<<cg, cb, 0, stream>>>(att_we, WeT, ENCd, ATTd);
  transpose_bf16_kernel<<<cg, cb, 0, stream>>>(out_w, OutWT, DECd + ENCd, Md);
  transpose_bf16_kernel<<<cg, cb, 0, stream>>>(pn_w1, PnW1T, Md, PHd);
  transpose_bf16_kernel<<<cg, cb, 0, stream>>>(pn_w2, PnW2T, PHd, POd);

  prenet_kernel<<<(Tt * Bb) / 16, 128, 143360, stream>>>(prevMelB, PnW1T,
                                                         PnW2T, pn_b1, pn_b2,
                                                         X);
  encproj_kernel<<<(Bb * Ss) / 32, 256, 163840, stream>>>(encB, WeT, att_be,
                                                          encProj);
  decoder_kernel<<<1, 1024, 213504, stream>>>(
      encB, encProj, X, WihB, WhhB, WdT, OutWT, att_bd, att_v, att_vb,
      gru_bih, gru_bhh, out_b, stop_w, stop_b, out);
}
